// AdaptiveTokenMerger_49460843381419
// MI455X (gfx1250) — matrix operand order repaired and checked by reading, not compiled, not run on hardware
//
#include <hip/hip_runtime.h>
#include <hip/hip_bf16.h>
#include <math.h>

#define BATCH 4
#define LSEQ  8192
#define DIM   512
#define HID   128
#define PFC   256
#define NTILES 8      // 128 / 16
#define KT1   16      // 512 / 32
#define KT2   24      // 768 / 32

typedef __attribute__((ext_vector_type(16))) __bf16 v16bf;
typedef __attribute__((ext_vector_type(8)))  float  v8f;

union BF16x16 {
  v16bf v;
  __bf16 h[16];
  unsigned short s[16];
};

__device__ __forceinline__ unsigned short f2bf(float f) {
  unsigned u = __float_as_uint(f);
  u += 0x7FFFu + ((u >> 16) & 1u);     // round-to-nearest-even
  return (unsigned short)(u >> 16);
}

// ---------------------------------------------------------------------------
// Pack fp32 weight matrix (K x 128, row-major) into bf16 in the exact per-lane
// WMMA B-fragment order: idx = ((ktile*8 + ntile)*32 + lane)*16 + i
//   element i of lane l -> B[ktile*32 + 16*(i>=8) + 8*(l>=16) + (i&7)][ntile*16 + (l&15)]
// ---------------------------------------------------------------------------
__global__ void pack_weights_kernel(const float* __restrict__ W,
                                    unsigned short* __restrict__ Wp, int KT) {
  const int idx = blockIdx.x * blockDim.x + threadIdx.x;
  const int total = KT * NTILES * 32 * 16;
  if (idx >= total) return;
  const int i     = idx & 15;
  const int lane  = (idx >> 4) & 31;
  const int ntile = (idx >> 9) & 7;
  const int ktile = idx >> 12;
  const int K = ktile * 32 + ((i >> 3) << 4) + ((lane >> 4) << 3) + (i & 7);
  const int n = ntile * 16 + (lane & 15);
  Wp[idx] = f2bf(W[K * HID + n]);
}

// ---------------------------------------------------------------------------
// Scorer: content MLP (WMMA GEMM1), grouped conv, boundary MLP (WMMA GEMM2),
// signal = content*(1-bscore)*mask, e = ||x*mask||^2.  64 threads (2 waves),
// 32 tokens per workgroup (16 per wave).
// ---------------------------------------------------------------------------
__launch_bounds__(64)
__global__ void scorer_kernel(const float* __restrict__ x, const float* __restrict__ am,
                              const float* __restrict__ b1, const float* __restrict__ W2,
                              const float* __restrict__ b2, const float* __restrict__ Wc,
                              const float* __restrict__ bc, const float* __restrict__ bb1,
                              const float* __restrict__ Wb2, const float* __restrict__ bb2,
                              const unsigned short* __restrict__ W1p,
                              const unsigned short* __restrict__ Wb1p,
                              float* __restrict__ signal_out, float* __restrict__ e_out) {
  __shared__ float pf_lds[32 * PFC];   // conv features, 32 KB
  __shared__ float h_lds[32 * HID];    // hidden activations, 16 KB
  __shared__ float cont_lds[32];

  const int wave    = threadIdx.x >> 5;
  const int lane    = threadIdx.x & 31;
  const int lrow    = lane & 15;       // M (A row) or N (B/C col)
  const int lhalf   = lane >> 4;       // K-interleave half
  const int tokBase = blockIdx.x * 32;
  const float* xrow = x + (size_t)(tokBase + wave * 16 + lrow) * DIM;

  v8f acc[NTILES];

  // ----- GEMM1: h1 = relu(x @ W1 + b1) -----
  #pragma unroll
  for (int nt = 0; nt < NTILES; ++nt)
    #pragma unroll
    for (int r = 0; r < 8; ++r) acc[nt][r] = 0.0f;

  for (int kt = 0; kt < KT1; ++kt) {
    BF16x16 a;
    const int kb = kt * 32 + lhalf * 8;
    __builtin_prefetch(xrow + kb + 64, 0, 0);      // global_prefetch_b8
    #pragma unroll
    for (int i = 0; i < 8; ++i) a.h[i]     = (__bf16)xrow[kb + i];       // v_cvt_pk_bf16_f32
    #pragma unroll
    for (int i = 0; i < 8; ++i) a.h[8 + i] = (__bf16)xrow[kb + 16 + i];
    const unsigned short* wp = W1p + (size_t)kt * (NTILES * 512) + lane * 16;
    #pragma unroll
    for (int nt = 0; nt < NTILES; ++nt) {
      BF16x16 bm;
      bm.v = *reinterpret_cast<const v16bf*>(wp + nt * 512);
      acc[nt] = __builtin_amdgcn_wmma_f32_16x16x32_bf16(
          false, a.v, false, bm.v, (short)0, acc[nt], false, false);
    }
  }
  #pragma unroll
  for (int nt = 0; nt < NTILES; ++nt) {
    const float bias = b1[nt * 16 + lrow];
    #pragma unroll
    for (int r = 0; r < 8; ++r) {
      float v = acc[nt][r] + bias;
      v = v > 0.0f ? v : 0.0f;
      const int tt = wave * 16 + lhalf * 8 + r;    // C layout: M = r + 8*lhalf
      h_lds[tt * HID + nt * 16 + lrow] = v;
    }
  }
  __syncthreads();

  // ----- content = sigmoid(h1 @ W2 + b2) -----
  if (threadIdx.x < 32) {
    const float* h = &h_lds[threadIdx.x * HID];
    float s = b2[0];
    for (int j = 0; j < HID; ++j) s += h[j] * W2[j];
    cont_lds[threadIdx.x] = 1.0f / (1.0f + expf(-s));
  }
  __syncthreads();

  // ----- grouped conv: pf[t][o], groups of 4 in-ch -> 2 out-ch, k=3, pad 1 -----
  for (int idx = threadIdx.x; idx < 32 * PFC; idx += blockDim.x) {
    const int tt   = idx >> 8;
    const int o    = idx & 255;
    const int g    = o >> 1;
    const int gtok = tokBase + tt;
    const int bb   = gtok / LSEQ;
    const int ll   = gtok - bb * LSEQ;
    float s = bc[o];
    const float* wc = Wc + o * 12;               // Wc[o][i][k], row-major
    #pragma unroll
    for (int k = 0; k < 3; ++k) {
      const int l2 = ll + k - 1;
      if (l2 < 0 || l2 >= LSEQ) continue;
      const float* xp = x + ((size_t)bb * LSEQ + l2) * DIM + g * 4;
      #pragma unroll
      for (int i = 0; i < 4; ++i) s += xp[i] * wc[i * 3 + k];
    }
    pf_lds[idx] = s;
  }
  __syncthreads();

  // ----- GEMM2: h2 = relu([x, pf] @ Wb1 + bb1), K = 512 (global) + 256 (LDS) -----
  #pragma unroll
  for (int nt = 0; nt < NTILES; ++nt)
    #pragma unroll
    for (int r = 0; r < 8; ++r) acc[nt][r] = 0.0f;

  const float* pfr = &pf_lds[(wave * 16 + lrow) * PFC];
  for (int kt = 0; kt < KT2; ++kt) {
    BF16x16 a;
    if (kt < KT1) {
      const int kb = kt * 32 + lhalf * 8;
      #pragma unroll
      for (int i = 0; i < 8; ++i) a.h[i]     = (__bf16)xrow[kb + i];
      #pragma unroll
      for (int i = 0; i < 8; ++i) a.h[8 + i] = (__bf16)xrow[kb + 16 + i];
    } else {
      const int kb = (kt - KT1) * 32 + lhalf * 8;
      #pragma unroll
      for (int i = 0; i < 8; ++i) a.h[i]     = (__bf16)pfr[kb + i];
      #pragma unroll
      for (int i = 0; i < 8; ++i) a.h[8 + i] = (__bf16)pfr[kb + 16 + i];
    }
    const unsigned short* wp = Wb1p + (size_t)kt * (NTILES * 512) + lane * 16;
    #pragma unroll
    for (int nt = 0; nt < NTILES; ++nt) {
      BF16x16 bm;
      bm.v = *reinterpret_cast<const v16bf*>(wp + nt * 512);
      acc[nt] = __builtin_amdgcn_wmma_f32_16x16x32_bf16(
          false, a.v, false, bm.v, (short)0, acc[nt], false, false);
    }
  }
  #pragma unroll
  for (int nt = 0; nt < NTILES; ++nt) {
    const float bias = bb1[nt * 16 + lrow];
    #pragma unroll
    for (int r = 0; r < 8; ++r) {
      float v = acc[nt][r] + bias;
      v = v > 0.0f ? v : 0.0f;
      const int tt = wave * 16 + lhalf * 8 + r;
      h_lds[tt * HID + nt * 16 + lrow] = v;
    }
  }
  __syncthreads();

  // ----- bscore, signal, e -----
  if (threadIdx.x < 32) {
    const int tt   = threadIdx.x;
    const int gtok = tokBase + tt;
    const float* h = &h_lds[tt * HID];
    float s = bb2[0];
    for (int j = 0; j < HID; ++j) s += h[j] * Wb2[j];
    const float bscore = 1.0f / (1.0f + expf(-s));
    const float mask   = am[gtok];
    signal_out[gtok] = cont_lds[tt] * (1.0f - bscore) * mask;
    const float* xr = x + (size_t)gtok * DIM;
    float es = 0.0f;
    for (int d = 0; d < DIM; ++d) { const float v = xr[d] * mask; es += v * v; }
    e_out[gtok] = es;
  }
}

// ---------------------------------------------------------------------------
// Exact 0.7-quantile per batch: order statistics via bit-binary-search over
// positive-float patterns (monotone), linear interpolation as jnp.quantile.
// ---------------------------------------------------------------------------
__global__ void quantile_kernel(const float* __restrict__ signal, float* __restrict__ thr) {
  __shared__ unsigned su[LSEQ];
  __shared__ int cnt;
  const int b = blockIdx.x;
  for (int i = threadIdx.x; i < LSEQ; i += blockDim.x)
    su[i] = __float_as_uint(signal[b * LSEQ + i]);
  __syncthreads();
  const float pos  = 0.7f * (float)(LSEQ - 1);
  const int   i0   = (int)floorf(pos);
  const float frac = pos - (float)i0;
  float vals[2];
  for (int which = 0; which < 2; ++which) {
    const int k = i0 + which;
    unsigned lo = 0u, hi = 0x7F800000u;
    while (lo < hi) {
      const unsigned mid = lo + ((hi - lo) >> 1);
      if (threadIdx.x == 0) cnt = 0;
      __syncthreads();
      int c = 0;
      for (int i = threadIdx.x; i < LSEQ; i += blockDim.x) c += (su[i] <= mid) ? 1 : 0;
      atomicAdd(&cnt, c);
      __syncthreads();
      const int total = cnt;
      __syncthreads();
      if (total >= k + 1) hi = mid; else lo = mid + 1;
    }
    vals[which] = __uint_as_float(lo);
  }
  if (threadIdx.x == 0) thr[b] = vals[0] + frac * (vals[1] - vals[0]);
}

// ---------------------------------------------------------------------------
// Serial boundary enforcement (carried-dependence scan), cumsum, seg_bounds.
// One thread per batch; cost ~4*8192 trivial iterations.
// ---------------------------------------------------------------------------
__global__ void enforce_kernel(const float* __restrict__ signal, const float* __restrict__ thr,
                               int* __restrict__ seg, float* __restrict__ seg_bounds,
                               int* __restrict__ accA, int* __restrict__ startA,
                               int* __restrict__ szA, int* __restrict__ kA) {
  if (threadIdx.x != 0) return;
  const int b = blockIdx.x;
  const float t = thr[b];
  const float* sg = signal + b * LSEQ;
  int* acc = accA + b * LSEQ;
  int* st  = startA + b * LSEQ;
  int* sza = szA + b * LSEQ;
  int* ka  = kA + b * LSEQ;

  int start = 0;
  for (int p = 0; p < LSEQ; ++p) {
    const bool cand = (sg[p] < t) || (p == LSEQ - 1);
    const int  ps   = p - start;
    const bool a    = cand && (ps >= 4);
    const int  kk   = a ? (ps + 15) / 16 : 1;
    int sz = a ? (ps / kk) : 1;
    if (sz < 1) sz = 1;
    acc[p] = a ? 1 : 0; st[p] = start; sza[p] = sz; ka[p] = kk;
    if (a) start = p;
  }
  int cs = 0, cz = 1, ck = 1, cv = 0;
  for (int p = LSEQ - 1; p >= 0; --p) {
    const int fs = cs, fz = cz, fk = ck, fv = cv;
    const int a = acc[p];
    if (a) { cs = st[p]; cz = sza[p]; ck = ka[p]; }
    cv |= a;
    const int r = p - fs;
    int nb = a;
    if (fv && r > 0 && (r % fz) == 0) {
      const int j = r / fz;
      if (j >= 1 && j <= fk - 1) nb = 1;
    }
    ka[p] = nb;                    // reuse as new_b
  }
  int s = 0;
  for (int p = 0; p < LSEQ; ++p) { s += ka[p]; seg[b * LSEQ + p] = s; }
  const int n_last = s;
  for (int p = 0; p < LSEQ; ++p)
    seg_bounds[b * LSEQ + p] = (p >= 1 && p <= n_last) ? 1.0f : 0.0f;
}

// ---------------------------------------------------------------------------
__global__ void init_kernel(float* __restrict__ merged, int* __restrict__ m,
                            float* __restrict__ denom) {
  const size_t total  = (size_t)BATCH * LSEQ * DIM;
  const size_t stride = (size_t)gridDim.x * blockDim.x;
  const size_t base   = (size_t)blockIdx.x * blockDim.x + threadIdx.x;
  for (size_t i = base; i < total; i += stride) merged[i] = 0.0f;
  for (size_t i = base; i < (size_t)BATCH * LSEQ; i += stride) {
    m[i] = (int)0xFF800000;   // -inf pattern; e >= 0 so signed-int max works
    denom[i] = 0.0f;
  }
}

__global__ void segmax_kernel(const float* __restrict__ e, const int* __restrict__ seg,
                              int* __restrict__ m) {
  const int t = blockIdx.x * blockDim.x + threadIdx.x;
  if (t >= BATCH * LSEQ) return;
  const int sf = seg[t] + (t / LSEQ) * LSEQ;
  atomicMax(&m[sf], __float_as_int(e[t]));
}

__global__ void segexp_kernel(const float* __restrict__ e, const int* __restrict__ seg,
                              const int* __restrict__ m, float* __restrict__ w,
                              float* __restrict__ denom) {
  const int t = blockIdx.x * blockDim.x + threadIdx.x;
  if (t >= BATCH * LSEQ) return;
  const int sf = seg[t] + (t / LSEQ) * LSEQ;
  const float ww = expf(e[t] - __int_as_float(m[sf]));
  w[t] = ww;
  atomicAdd(&denom[sf], ww);
}

__global__ void merge_kernel(const float* __restrict__ x, const float* __restrict__ am,
                             const int* __restrict__ seg, const float* __restrict__ w,
                             const float* __restrict__ denom, float* __restrict__ merged) {
  const int t  = blockIdx.x;
  const int sf = seg[t] + (t / LSEQ) * LSEQ;
  const float scale = am[t] * (w[t] / denom[sf]);
  const float* xr = x + (size_t)t * DIM;
  float* out = merged + (size_t)sf * DIM;
  for (int d = threadIdx.x; d < DIM; d += blockDim.x)
    atomicAdd(&out[d], xr[d] * scale);
}

// ---------------------------------------------------------------------------
extern "C" void kernel_launch(void* const* d_in, const int* in_sizes, int n_in,
                              void* d_out, int out_size, void* d_ws, size_t ws_size,
                              hipStream_t stream) {
  const float* x   = (const float*)d_in[0];
  const float* am  = (const float*)d_in[1];
  const float* W1  = (const float*)d_in[2];
  const float* b1  = (const float*)d_in[3];
  const float* W2  = (const float*)d_in[4];
  const float* b2  = (const float*)d_in[5];
  const float* Wc  = (const float*)d_in[6];
  const float* bc  = (const float*)d_in[7];
  const float* Wb1 = (const float*)d_in[8];
  const float* bb1 = (const float*)d_in[9];
  const float* Wb2 = (const float*)d_in[10];
  const float* bb2 = (const float*)d_in[11];

  char* ws = (char*)d_ws;
  unsigned short* W1p  = (unsigned short*)(ws + 0);         // 128 KB
  unsigned short* Wb1p = (unsigned short*)(ws + 131072);    // 192 KB
  float* signal = (float*)(ws + 327680);
  float* e      = (float*)(ws + 458752);
  float* thr    = (float*)(ws + 589824);
  int*   seg    = (int*)  (ws + 590080);
  int*   m      = (int*)  (ws + 721152);
  float* denom  = (float*)(ws + 852224);
  float* w      = (float*)(ws + 983296);
  int*   accA   = (int*)  (ws + 1114368);
  int*   startA = (int*)  (ws + 1245440);
  int*   szA    = (int*)  (ws + 1376512);
  int*   kA     = (int*)  (ws + 1507584);

  float* merged     = (float*)d_out;
  float* seg_bounds = merged + (size_t)BATCH * LSEQ * DIM;

  pack_weights_kernel<<<(KT1 * NTILES * 32 * 16) / 256, 256, 0, stream>>>(W1, W1p, KT1);
  pack_weights_kernel<<<(KT2 * NTILES * 32 * 16) / 256, 256, 0, stream>>>(Wb1, Wb1p, KT2);
  init_kernel<<<2048, 256, 0, stream>>>(merged, m, denom);
  scorer_kernel<<<(BATCH * LSEQ) / 32, 64, 0, stream>>>(
      x, am, b1, W2, b2, Wc, bc, bb1, Wb2, bb2, W1p, Wb1p, signal, e);
  quantile_kernel<<<BATCH, 256, 0, stream>>>(signal, thr);
  enforce_kernel<<<BATCH, 32, 0, stream>>>(signal, thr, seg, seg_bounds,
                                           accA, startA, szA, kA);
  segmax_kernel<<<(BATCH * LSEQ) / 256, 256, 0, stream>>>(e, seg, m);
  segexp_kernel<<<(BATCH * LSEQ) / 256, 256, 0, stream>>>(e, seg, m, w, denom);
  merge_kernel<<<BATCH * LSEQ, 256, 0, stream>>>(x, am, seg, w, denom, merged);
}
